// CELPNetSub_18305150616354
// MI455X (gfx1250) — matrix operand order repaired and checked by reading, not compiled, not run on hardware
//
#include <hip/hip_runtime.h>
#include <hip/hip_bf16.h>

// ---------------------------------------------------------------------------
// CELPNet sub-block fused kernel for gfx1250 (MI455X), bf16 WMMA 16x16x32.
// ---------------------------------------------------------------------------

typedef __attribute__((ext_vector_type(8)))  __bf16 v8bf;
typedef __attribute__((ext_vector_type(16))) __bf16 v16bf;
typedef __attribute__((ext_vector_type(8)))  float  v8f;

#define BDIM 16384
#define SDIM 40
#define CDIM 256
#define K1PAD 320      // concat(cond[256], prev[40]) padded 296 -> 320
#define ASTRIDE 320    // LDS activation row stride (bf16 elements)

// ---- fast transcendentals: branchless, hardware ops ----
__device__ __forceinline__ float fast_exp2(float x) {
#if __has_builtin(__builtin_amdgcn_exp2f)
    return __builtin_amdgcn_exp2f(x);
#else
    return exp2f(x);
#endif
}
__device__ __forceinline__ float fast_rcp(float x) {
#if __has_builtin(__builtin_amdgcn_rcpf)
    return __builtin_amdgcn_rcpf(x);
#else
    return 1.0f / x;
#endif
}
__device__ __forceinline__ float fast_tanh(float x) {
#if __has_builtin(__builtin_amdgcn_tanhf)
    return __builtin_amdgcn_tanhf(x);        // v_tanh_f32 (CDNA5 trans op)
#else
    // tanh(x) = 1 - 2/(1 + 2^(2*log2(e)*x)); inf-safe at both tails
    return 1.0f - 2.0f * fast_rcp(1.0f + fast_exp2(2.88539008177793f * x));
#endif
}
__device__ __forceinline__ float fast_sigmoid(float x) {
    return 0.5f + 0.5f * fast_tanh(0.5f * x);
}

__device__ __forceinline__ unsigned short f32_to_bf16(float f) {
    __bf16 h = (__bf16)f;                    // v_cvt_bf16_f32
    return __builtin_bit_cast(unsigned short, h);
}

// A-fragment (16x32 bf16, row-major in LDS). ISA layout: lanes 0-15 hold row
// (lane&15) with K {kt*32+0..7, kt*32+16..23}; lanes 16-31 hold K {+8..15,+24..31}.
__device__ __forceinline__ v16bf load_a_frag(const unsigned short* s, int stride,
                                             int kt, int lane) {
    const int row = lane & 15;
    const int hlf = lane >> 4;
    const unsigned short* p = s + row * stride + kt * 32 + hlf * 8;
    v8bf lo = *(const v8bf*)p;          // K +0..7  (or +8..15)
    v8bf hi = *(const v8bf*)(p + 16);   // K +16..23 (or +24..31)
    v16bf a;
#pragma unroll
    for (int j = 0; j < 8; ++j) { a[j] = lo[j]; a[j + 8] = hi[j]; }
    return a;
}

// B-fragment: pre-swizzled by pack_weight; 16 contiguous bf16 per lane.
__device__ __forceinline__ v16bf load_b_frag(const unsigned short* w, int tile,
                                             int lane) {
    return *(const v16bf*)(w + (size_t)tile * 512 + (size_t)lane * 16);
}

__device__ __forceinline__ v8f wmma_bf16(v16bf a, v16bf b, v8f c) {
    return __builtin_amdgcn_wmma_f32_16x16x32_bf16(
        /*neg_a=*/false, a, /*neg_b=*/false, b,
        /*c_mod=*/(short)0, c, /*reuse_a=*/false, /*reuse_b=*/false);
}

// ---------------------------------------------------------------------------
// Weight pre-pack: W is [N,K] f32 (PyTorch layout, used as x @ W^T).
// Effective B matrix is W^T [K,N]. For each (nt,kt) 32x16 tile, lane L holds
// column n = nt*16 + (L&15), K rows kt*32 + (L>>4)*8 + (j&7) + (j>>3)*16, j=0..15 (bf16).
// Zero-padded outside [N,K].
// ---------------------------------------------------------------------------
__global__ void pack_weight(const float* __restrict__ W, unsigned short* __restrict__ dst,
                            int N, int K, int KT, int total) {
    int idx = blockIdx.x * blockDim.x + threadIdx.x;
    if (idx >= total) return;
    int t = idx >> 9;
    int l = (idx >> 4) & 31;
    int j = idx & 15;
    int kt = t % KT;
    int nt = t / KT;
    int n = nt * 16 + (l & 15);
    int k = kt * 32 + ((l >> 4) * 8) + (j & 7) + ((j >> 3) * 16);
    float v = (n < N && k < K) ? W[(size_t)n * K + k] : 0.0f;
    dst[idx] = f32_to_bf16(v);
}

// ---------------------------------------------------------------------------
// Dense layer: dst = tanh(src[16xK] @ W^T + b); each wave owns 2 column tiles.
// A fragments loaded once per kt and shared by both tiles.
// ---------------------------------------------------------------------------
__device__ __forceinline__ void dense_layer(
    const unsigned short* src, unsigned short* dst,
    const unsigned short* __restrict__ pw, const float* __restrict__ bias,
    int KT, int lane, int wave, int hlf) {
    const int nt0 = wave * 2;
    v8f acc0 = {}, acc1 = {};
    for (int kt = 0; kt < KT; ++kt) {
        v16bf a = load_a_frag(src, ASTRIDE, kt, lane);
        acc0 = wmma_bf16(a, load_b_frag(pw, (nt0    ) * KT + kt, lane), acc0);
        acc1 = wmma_bf16(a, load_b_frag(pw, (nt0 + 1) * KT + kt, lane), acc1);
    }
    const int c0 = nt0 * 16 + (lane & 15);
    const float bias0 = bias[c0], bias1 = bias[c0 + 16];
#pragma unroll
    for (int r = 0; r < 8; ++r) {
        const int m = r + 8 * hlf;
        dst[m * ASTRIDE + c0]      = f32_to_bf16(fast_tanh(acc0[r] + bias0));
        dst[m * ASTRIDE + c0 + 16] = f32_to_bf16(fast_tanh(acc1[r] + bias1));
    }
}

// ---------------------------------------------------------------------------
// One GRU layer for a 16-row tile, gates fully in registers.
// A fragments (x and h) loaded once per kt, feeding 6 WMMAs each.
// ---------------------------------------------------------------------------
__device__ __forceinline__ void gru_layer(
    const unsigned short* actIn, unsigned short* actOut, unsigned short* sH,
    const float* __restrict__ hglob,
    const float* __restrict__ bih, const float* __restrict__ bhh,
    const unsigned short* __restrict__ pwih, const unsigned short* __restrict__ pwhh,
    float* __restrict__ gout,
    int rowBase, int tid, int lane, int wave, int hlf) {
    // stage h (bf16, row-major) into LDS
    for (int i = tid; i < 16 * CDIM; i += 256)
        sH[i] = f32_to_bf16(hglob[(size_t)(rowBase + (i >> 8)) * CDIM + (i & 255)]);
    __syncthreads();

    const int KT = 8;
    const int ct0 = wave * 2;              // first column tile within C (0..15)
    v8f acc[2][6];
#pragma unroll
    for (int s = 0; s < 2; ++s)
#pragma unroll
        for (int g = 0; g < 6; ++g) acc[s][g] = (v8f){};

    for (int kt = 0; kt < KT; ++kt) {
        v16bf ax = load_a_frag(actIn, ASTRIDE, kt, lane);
        v16bf ah = load_a_frag(sH, CDIM, kt, lane);
#pragma unroll
        for (int s = 0; s < 2; ++s) {
            const int ct = ct0 + s;
            acc[s][0] = wmma_bf16(ax, load_b_frag(pwih, (ct     ) * KT + kt, lane), acc[s][0]);
            acc[s][1] = wmma_bf16(ax, load_b_frag(pwih, (ct + 16) * KT + kt, lane), acc[s][1]);
            acc[s][2] = wmma_bf16(ax, load_b_frag(pwih, (ct + 32) * KT + kt, lane), acc[s][2]);
            acc[s][3] = wmma_bf16(ah, load_b_frag(pwhh, (ct     ) * KT + kt, lane), acc[s][3]);
            acc[s][4] = wmma_bf16(ah, load_b_frag(pwhh, (ct + 16) * KT + kt, lane), acc[s][4]);
            acc[s][5] = wmma_bf16(ah, load_b_frag(pwhh, (ct + 32) * KT + kt, lane), acc[s][5]);
        }
    }

#pragma unroll
    for (int s = 0; s < 2; ++s) {
        const int col = (ct0 + s) * 16 + (lane & 15);
        const float bir = bih[col], biz = bih[col + 256], bin = bih[col + 512];
        const float bhr = bhh[col], bhz = bhh[col + 256], bhn = bhh[col + 512];
#pragma unroll
        for (int r = 0; r < 8; ++r) {
            const int m = r + 8 * hlf;
            float rg = fast_sigmoid(acc[s][0][r] + bir + acc[s][3][r] + bhr);
            float zg = fast_sigmoid(acc[s][1][r] + biz + acc[s][4][r] + bhz);
            float ng = fast_tanh(acc[s][2][r] + bin + rg * (acc[s][5][r] + bhn));
            float hv = hglob[(size_t)(rowBase + m) * CDIM + col];
            float nh = (1.0f - zg) * ng + zg * hv;
            gout[(size_t)(rowBase + m) * CDIM + col] = nh;
            actOut[m * ASTRIDE + col] = f32_to_bf16(nh);
        }
    }
    __syncthreads();
}

// ---------------------------------------------------------------------------
// Fused kernel: one workgroup = 16 batch rows through the whole network.
// ---------------------------------------------------------------------------
__global__ __launch_bounds__(256)
void celpnet_fused(
    const float* __restrict__ cond, const float* __restrict__ prev,
    const float* __restrict__ h1, const float* __restrict__ h2, const float* __restrict__ h3,
    const float* __restrict__ b_d1, const float* __restrict__ b_d2,
    const float* __restrict__ b_ih1, const float* __restrict__ b_hh1,
    const float* __restrict__ b_ih2, const float* __restrict__ b_hh2,
    const float* __restrict__ b_ih3, const float* __restrict__ b_hh3,
    const float* __restrict__ b_out,
    const unsigned short* __restrict__ pw_d1, const unsigned short* __restrict__ pw_d2,
    const unsigned short* __restrict__ pw_ih1, const unsigned short* __restrict__ pw_hh1,
    const unsigned short* __restrict__ pw_ih2, const unsigned short* __restrict__ pw_hh2,
    const unsigned short* __restrict__ pw_ih3, const unsigned short* __restrict__ pw_hh3,
    const unsigned short* __restrict__ pw_out,
    float* __restrict__ out) {
    __shared__ unsigned short sX[16 * ASTRIDE];
    __shared__ unsigned short sY[16 * ASTRIDE];
    __shared__ unsigned short sH[16 * CDIM];

    const int tid  = threadIdx.x;
    const int lane = tid & 31;
    const int wave = tid >> 5;
    const int hlf  = lane >> 4;
    const int rowBase = blockIdx.x * 16;

    float* g1o = out + (size_t)BDIM * SDIM;
    float* g2o = g1o + (size_t)BDIM * CDIM;
    float* g3o = g2o + (size_t)BDIM * CDIM;

    // ---- stage concat(cond, prev) as bf16, zero-padded to 320 ----
    for (int i = tid; i < 16 * K1PAD; i += 256) {
        int r = i / K1PAD, c = i % K1PAD;
        float v;
        if (c < CDIM)             v = cond[(size_t)(rowBase + r) * CDIM + c];
        else if (c < CDIM + SDIM) v = prev[(size_t)(rowBase + r) * SDIM + (c - CDIM)];
        else                      v = 0.0f;
        sX[i] = f32_to_bf16(v);
    }
    __syncthreads();

    // ---- dense 1: sY = tanh(sX[16x320] @ w_d1^T + b_d1) ----
    dense_layer(sX, sY, pw_d1, b_d1, K1PAD / 32, lane, wave, hlf);
    __syncthreads();

    // ---- dense 2: sX = tanh(sY[16x256] @ w_d2^T + b_d2) ----
    dense_layer(sY, sX, pw_d2, b_d2, 8, lane, wave, hlf);
    __syncthreads();

    // ---- 3 stacked GRU cells (x: sX -> sY -> sX -> sY) ----
    gru_layer(sX, sY, sH, h1, b_ih1, b_hh1, pw_ih1, pw_hh1, g1o, rowBase, tid, lane, wave, hlf);
    gru_layer(sY, sX, sH, h2, b_ih2, b_hh2, pw_ih2, pw_hh2, g2o, rowBase, tid, lane, wave, hlf);
    gru_layer(sX, sY, sH, h3, b_ih3, b_hh3, pw_ih3, pw_hh3, g3o, rowBase, tid, lane, wave, hlf);

    // ---- output: out = tanh(g3[16x256] @ w_out^T + b_out), N padded 40->48 ----
    if (wave < 3) {
        int nt = wave;
        v8f acc = {};
        for (int kt = 0; kt < 8; ++kt) {
            v16bf a = load_a_frag(sY, ASTRIDE, kt, lane);
            v16bf b = load_b_frag(pw_out, nt * 8 + kt, lane);
            acc = wmma_bf16(a, b, acc);
        }
        int col = nt * 16 + (lane & 15);
        if (col < SDIM) {
            float bias = b_out[col];
#pragma unroll
            for (int r = 0; r < 8; ++r)
                out[(size_t)(rowBase + r + 8 * hlf) * SDIM + col] = fast_tanh(acc[r] + bias);
        }
    }
}

// ---------------------------------------------------------------------------
// Host-side launch
// ---------------------------------------------------------------------------
extern "C" void kernel_launch(void* const* d_in, const int* in_sizes, int n_in,
                              void* d_out, int out_size, void* d_ws, size_t ws_size,
                              hipStream_t stream) {
    (void)in_sizes; (void)n_in; (void)out_size; (void)ws_size;

    const float* cond  = (const float*)d_in[0];
    const float* prev  = (const float*)d_in[1];
    const float* h1    = (const float*)d_in[2];
    const float* h2    = (const float*)d_in[3];
    const float* h3    = (const float*)d_in[4];
    const float* w_d1  = (const float*)d_in[5];
    const float* b_d1  = (const float*)d_in[6];
    const float* w_d2  = (const float*)d_in[7];
    const float* b_d2  = (const float*)d_in[8];
    const float* w_ih1 = (const float*)d_in[9];
    const float* w_hh1 = (const float*)d_in[10];
    const float* b_ih1 = (const float*)d_in[11];
    const float* b_hh1 = (const float*)d_in[12];
    const float* w_ih2 = (const float*)d_in[13];
    const float* w_hh2 = (const float*)d_in[14];
    const float* b_ih2 = (const float*)d_in[15];
    const float* b_hh2 = (const float*)d_in[16];
    const float* w_ih3 = (const float*)d_in[17];
    const float* w_hh3 = (const float*)d_in[18];
    const float* b_ih3 = (const float*)d_in[19];
    const float* b_hh3 = (const float*)d_in[20];
    const float* w_out = (const float*)d_in[21];
    const float* b_out = (const float*)d_in[22];
    float* out = (float*)d_out;

    // Packed-weight workspace layout (bf16 elements)
    unsigned short* ws = (unsigned short*)d_ws;
    const size_t n_d1 = 16 * 10 * 512;  // N=256 (16 tiles), Kpad=320 (10 tiles)
    const size_t n_d2 = 16 * 8 * 512;   // N=256, K=256
    const size_t n_g  = 48 * 8 * 512;   // N=768, K=256
    const size_t n_o  = 3 * 8 * 512;    // Npad=48, K=256

    unsigned short* p_d1  = ws;
    unsigned short* p_d2  = p_d1 + n_d1;
    unsigned short* p_ih1 = p_d2 + n_d2;
    unsigned short* p_hh1 = p_ih1 + n_g;
    unsigned short* p_ih2 = p_hh1 + n_g;
    unsigned short* p_hh2 = p_ih2 + n_g;
    unsigned short* p_ih3 = p_hh2 + n_g;
    unsigned short* p_hh3 = p_ih3 + n_g;
    unsigned short* p_out = p_hh3 + n_g;

    auto pack = [&](const float* W, unsigned short* dst, int N, int K, int KT, size_t total) {
        int blocks = (int)((total + 255) / 256);
        pack_weight<<<blocks, 256, 0, stream>>>(W, dst, N, K, KT, (int)total);
    };
    pack(w_d1,  p_d1,  256, 296, 10, n_d1);
    pack(w_d2,  p_d2,  256, 256, 8,  n_d2);
    pack(w_ih1, p_ih1, 768, 256, 8,  n_g);
    pack(w_hh1, p_hh1, 768, 256, 8,  n_g);
    pack(w_ih2, p_ih2, 768, 256, 8,  n_g);
    pack(w_hh2, p_hh2, 768, 256, 8,  n_g);
    pack(w_ih3, p_ih3, 768, 256, 8,  n_g);
    pack(w_hh3, p_hh3, 768, 256, 8,  n_g);
    pack(w_out, p_out, 40,  256, 8,  n_o);

    celpnet_fused<<<BDIM / 16, 256, 0, stream>>>(
        cond, prev, h1, h2, h3,
        b_d1, b_d2, b_ih1, b_hh1, b_ih2, b_hh2, b_ih3, b_hh3, b_out,
        p_d1, p_d2, p_ih1, p_hh1, p_ih2, p_hh2, p_ih3, p_hh3, p_out,
        out);
}
